// NVAN_65257733095783
// MI455X (gfx1250) — compile-verified, matrix operand order checked by reading, not run on hardware
//
#include <hip/hip_runtime.h>
#include <math.h>

// ---------------- types for WMMA ----------------
typedef __bf16 v16bf __attribute__((ext_vector_type(16)));
typedef float  v8f   __attribute__((ext_vector_type(8)));

union U16BF { unsigned int u[8]; v16bf v; };

__device__ __forceinline__ unsigned short f2bf(float f) {
  unsigned int u = __float_as_uint(f);
  unsigned int r = u + 0x7FFFu + ((u >> 16) & 1u);
  return (unsigned short)(r >> 16);
}
__device__ __forceinline__ float sigf(float x) { return 1.0f / (1.0f + expf(-x)); }

// Model constants
#define NB 32     // B (scan length of the LSTM)
#define NT 128    // T (per-step batch of the LSTM)
#define NV 16     // V channels
#define NH 256    // H
#define ND 512    // D = 2H
#define NG 1024   // 4H
#define NW 385    // W = D-K+1
#define NK 128    // K
#define TM1 127   // T-1
#define FEPS 1e-5f
#define FSHARP 2.0f

// ================================================================
// Prep: build bf16 concat weights [Wih|Whh] (B-operand row-major) and bias sums
// ================================================================
__global__ void k_prep(const float* whh0f, const float* whh0b,
                       const float* wih1f, const float* whh1f,
                       const float* wih1b, const float* whh1b,
                       const float* bih0f, const float* bhh0f,
                       const float* bih0b, const float* bhh0b,
                       const float* bih1f, const float* bhh1f,
                       const float* bih1b, const float* bhh1b,
                       unsigned short* wcat0, unsigned short* wcat1,
                       float* bsum0, float* bsum1) {
  int i = blockIdx.x * blockDim.x + threadIdx.x;   // covers 2*1024*768
  if (i < 2 * 1024 * 768) {                        // wcat1: [dir][g][k], k<512 Wih else Whh
    int dir = i / (1024 * 768);
    int r = i % (1024 * 768);
    int g = r / 768, k = r % 768;
    float val;
    if (k < 512) val = (dir ? wih1b : wih1f)[g * 512 + k];
    else         val = (dir ? whh1b : whh1f)[g * 256 + (k - 512)];
    wcat1[i] = f2bf(val);
  }
  if (i < 2 * 1024 * 256) {                        // wcat0: [dir][g][k] = Whh
    int dir = i / (1024 * 256);
    int r = i % (1024 * 256);
    int g = r >> 8, k = r & 255;
    wcat0[i] = f2bf((dir ? whh0b : whh0f)[g * 256 + k]);
  }
  if (i < 2048) {
    int dir = i >> 10, g = i & 1023;
    bsum0[i] = (dir ? bih0b : bih0f)[g] + (dir ? bhh0b : bhh0f)[g];
    bsum1[i] = (dir ? bih1b : bih1f)[g] + (dir ? bhh1b : bhh1f)[g];
  }
}

// ================================================================
// Persistent bidirectional LSTM layer, fully fused.
// One workgroup per (v,dir); 16 waves. Each wave owns 8 fixed 16x16 patches
// of the (128 x 256) h/c grid: m0 = (wave&7)*16, j0 = ((wave>>3)+2i)*16.
// For each patch it accumulates the 4 gate tiles (cols q*256+j0) with WMMA,
// applies the LSTM nonlinearity in registers, and writes h to a
// double-buffered LDS tile + global. One barrier per step, no fences,
// no gate scratch in global memory. LDS: 2x64KB h (bf16) + 128KB c = 256KB.
// KX = input-projection K (0 for layer0 din=1, 512 for layer1).
// ================================================================
template <int KX>
__global__ __launch_bounds__(512) void k_lstm(
    const float* x0,              // layer0: raw x (B,T,V)
    const unsigned short* xbf,    // layer1: bf16 input (V,B,T,512)
    const unsigned short* wcat,   // (2,1024,KX+256) bf16 row-major
    const float* bsum,            // (2,1024) bih+bhh
    const float* wih_f, const float* wih_b, // layer0 rank-1 weights (1024)
    unsigned short* hout_bf,      // (V,B,T,512) bf16
    float* hout_f32)              // same layout f32, or nullptr
{
  const int K = KX + 256;
  const int v = blockIdx.x >> 1;
  const int dir = blockIdx.x & 1;
  const int tid = threadIdx.x;
  const int lane = tid & 31;
  const int wave = tid >> 5;
  const int mlane = lane & 15;
  const int kgrp = lane >> 4;
  const int m0 = (wave & 7) << 4;     // fixed row group of this wave
  const int j0base = wave >> 3;       // j0 index = j0base + 2*i

  __shared__ unsigned short hA[2][NT * NH];  // double-buffered h, bf16, [n][j]
  __shared__ float cS[NT * NH];              // cell state

  for (int i = tid; i < NT * NH; i += 512) { hA[0][i] = 0; cS[i] = 0.0f; }
  __syncthreads();

  const unsigned short* wd = wcat + (size_t)dir * 1024 * K;
  const float* bs = bsum + dir * 1024;
  const float* wih = dir ? wih_b : wih_f;

  for (int t = 0; t < NB; ++t) {
    const int s = dir ? (NB - 1 - t) : t;
    const unsigned short* hcur = hA[t & 1];
    unsigned short* hnxt = hA[(t + 1) & 1];
    const int mA = m0 + mlane;                     // A-operand row for this lane
    const size_t xrow = (((size_t)v * NB + s) * NT + mA) * ND;

    for (int i = 0; i < 8; ++i) {
      const int j0c = (j0base + 2 * i) << 4;       // column base in [0,256)
      v8f acc0 = {0.f,0.f,0.f,0.f,0.f,0.f,0.f,0.f};
      v8f acc1 = acc0, acc2 = acc0, acc3 = acc0;
      const size_t wrow = (size_t)(j0c + mlane) * K;   // B row (gate q=0)

      for (int k0 = 0; k0 < K; k0 += 32) {
        U16BF a, b0, b1, b2, b3;
#pragma unroll
        for (int e = 0; e < 8; ++e) {
          const int kp = (e < 4 ? 2 * e : 16 + 2 * (e - 4)) + kgrp * 8;
          const int kk = k0 + kp;
          if (KX > 0 && kk < KX) {
            a.u[e] = *(const unsigned int*)(xbf + xrow + kk);
          } else {
            a.u[e] = *(const unsigned int*)(&hcur[mA * NH + (kk - KX)]);
          }
          b0.u[e] = *(const unsigned int*)(wd + wrow + kk);
          b1.u[e] = *(const unsigned int*)(wd + wrow + (size_t)256 * K + kk);
          b2.u[e] = *(const unsigned int*)(wd + wrow + (size_t)512 * K + kk);
          b3.u[e] = *(const unsigned int*)(wd + wrow + (size_t)768 * K + kk);
        }
        acc0 = __builtin_amdgcn_wmma_f32_16x16x32_bf16(false, a.v, false, b0.v, (short)0, acc0, false, false);
        acc1 = __builtin_amdgcn_wmma_f32_16x16x32_bf16(false, a.v, false, b1.v, (short)0, acc1, false, false);
        acc2 = __builtin_amdgcn_wmma_f32_16x16x32_bf16(false, a.v, false, b2.v, (short)0, acc2, false, false);
        acc3 = __builtin_amdgcn_wmma_f32_16x16x32_bf16(false, a.v, false, b3.v, (short)0, acc3, false, false);
      }

      // ---- fused LSTM nonlinearity on the 16x16 patch (in registers) ----
#pragma unroll
      for (int r = 0; r < 8; ++r) {
        const int m = m0 + kgrp * 8 + r;           // C/D row for this slot
        const int j = j0c + mlane;                 // column within H
        float gi = acc0[r] + bs[j];
        float gf = acc1[r] + bs[256 + j];
        float gg = acc2[r] + bs[512 + j];
        float go = acc3[r] + bs[768 + j];
        if (KX == 0) {   // layer0: din==1 rank-1 input projection
          const float xv = x0[((size_t)s * NT + m) * NV + v];
          gi += xv * wih[j];
          gf += xv * wih[256 + j];
          gg += xv * wih[512 + j];
          go += xv * wih[768 + j];
        }
        const int cidx = m * NH + j;
        const float c = sigf(gf) * cS[cidx] + sigf(gi) * tanhf(gg);
        cS[cidx] = c;
        const float h = sigf(go) * tanhf(c);
        const unsigned short hb = f2bf(h);
        hnxt[cidx] = hb;
        const size_t oidx = (((size_t)v * NB + s) * NT + m) * ND + dir * NH + j;
        hout_bf[oidx] = hb;
        if (hout_f32) hout_f32[oidx] = h;
      }
    }
    __syncthreads();   // hnxt complete before it becomes hcur
  }
}

// ================================================================
// Tail kernels (hm layout: hm[v][b][t][d], f32)
// ================================================================
__global__ void k_ht(const float* hm, const float* sc_t_w, const float* sc_t_b,
                     const float* cc1_w, const float* cc1_b,
                     float* h_t, float* h_tit) {
  __shared__ float row[NV * ND];
  const int b = blockIdx.x, tid = threadIdx.x;
  for (int i = tid; i < NV * ND; i += 128) {
    const int vv = i >> 9, d = i & 511;
    row[i] = hm[(((size_t)vv * NB + b) * NT + (NT - 1)) * ND + d];
  }
  __syncthreads();
  const int k = tid;
  float bsumt = 0.f;
  for (int v2 = 0; v2 < NV; ++v2) bsumt += sc_t_b[v2];
  float a1 = bsumt, a2 = cc1_b[0];
  for (int vv = 0; vv < NV; ++vv)
    for (int w = 0; w < NW; ++w) {
      const float r = row[vv * ND + k + w];
      a1 += r * sc_t_w[vv * NW + w];
      a2 += r * cc1_w[vv * NW + w];
    }
  h_t[b * NK + k] = a1;
  h_tit[b * NK + k] = a2;
}

__global__ void k_hi(const float* hm, const float* sc_i_w, const float* sc_i_b,
                     const float* cc2_w, const float* cc2_b, const float* h_tit,
                     float* h_i, float* h_ti) {
  __shared__ float row[NV * ND];
  const int b = blockIdx.x / TM1;
  const int t = blockIdx.x % TM1;
  const int tid = threadIdx.x;
  for (int i = tid; i < NV * ND; i += 128) {
    const int vv = i >> 9, d = i & 511;
    row[i] = hm[(((size_t)vv * NB + b) * NT + t) * ND + d];
  }
  __syncthreads();
  const int k = tid;
  float bsumi = 0.f;
  for (int v2 = 0; v2 < NV; ++v2) bsumi += sc_i_b[v2];
  float a1 = bsumi, a2 = cc2_b[0];
  for (int vv = 0; vv < NV; ++vv)
    for (int w = 0; w < NW; ++w) {
      const float r = row[vv * ND + k + w];
      a1 += r * sc_i_w[vv * NW + w];
      a2 += r * cc2_w[vv * NW + w];
    }
  const size_t o = ((size_t)b * TM1 + t) * NK + k;
  h_i[o] = a1;
  h_ti[o] = h_tit[b * NK + k] + a2;
}

// attention scores aw[b,t,v] + softmax over v. One wave per (b,t).
__global__ void k_awsoft(const float* hm, float* soft) {
  const int gw = blockIdx.x * 4 + (threadIdx.x >> 5);
  const int lane = threadIdx.x & 31;
  if (gw >= NB * TM1) return;
  const int b = gw / TM1, t = gw % TM1;
  const int it = (t == 0) ? 0 : (t - 1);
  float aw[NV];
  for (int v = 0; v < NV; ++v) {
    float p = 0.f;
    const size_t base_i = (((size_t)v * NB + b) * NT + it) * ND;
    const size_t base_l = (((size_t)v * NB + b) * NT + (NT - 1)) * ND;
    for (int d = lane; d < ND; d += 32) p += hm[base_i + d] * hm[base_l + d];
    for (int o = 16; o > 0; o >>= 1) p += __shfl_xor(p, o);
    aw[v] = p;
  }
  float mx = aw[0];
  for (int v = 1; v < NV; ++v) mx = fmaxf(mx, aw[v]);
  float sum = 0.f;
  for (int v = 0; v < NV; ++v) sum += expf(aw[v] - mx);
  if (lane < NV)
    soft[((size_t)b * TM1 + t) * NV + lane] = expf(aw[lane] - mx) / sum;
}

__global__ void k_e(const float* h_t, const float* h_i, const float* h_ti,
                    const float* soft,
                    const float* wa_w, const float* wa_b,
                    const float* wb_w, const float* wb_b,
                    const float* wc_w, const float* wc_b,
                    const float* wd_w, const float* wd_b,
                    float* es) {
  const int id = blockIdx.x * blockDim.x + threadIdx.x;
  if (id >= NB * TM1 * NV) return;
  const int b = id / (TM1 * NV);
  const int r = id % (TM1 * NV);
  const int t = r / NV, vo = r % NV;
  float acc = wa_b[vo] + wb_b[vo] + wc_b[vo] + wd_b[vo];
  const float* ht = h_t + b * NK;
  const float* hi = h_i + ((size_t)b * TM1 + t) * NK;
  const float* hti = h_ti + ((size_t)b * TM1 + t) * NK;
  for (int k = 0; k < NK; ++k) {
    acc += wa_w[vo * NK + k] * ht[k];
    acc += wb_w[vo * NK + k] * hi[k];
    acc += wc_w[vo * NK + k] * hti[k];
  }
  const float* sf = soft + ((size_t)b * TM1 + t) * NV;
  for (int j = 0; j < NV; ++j) acc += wd_w[vo * NV + j] * sf[j];
  es[((size_t)b * NV + vo) * TM1 + t] = sigf(tanhf(acc));  // E transposed, sigmoid
}

__global__ void k_mm1(const float* es, float* rmin1, float* rmax1) {
  const int id = blockIdx.x * blockDim.x + threadIdx.x;   // (b,t)
  if (id >= NB * TM1) return;
  const int b = id / TM1, t = id % TM1;
  float mn = 1e30f, mx = -1e30f;
  for (int vo = 0; vo < NV; ++vo) {
    const float e = es[((size_t)b * NV + vo) * TM1 + t];
    mn = fminf(mn, e); mx = fmaxf(mx, e);
  }
  rmin1[id] = mn; rmax1[id] = mx;
}

__global__ void k_mm2(const float* es, float* rmin2, float* rmax2) {
  const int id = blockIdx.x * blockDim.x + threadIdx.x;   // (b,v)
  if (id >= NB * NV) return;
  float mn = 1e30f, mx = -1e30f;
  for (int t = 0; t < TM1; ++t) {
    const float e = es[(size_t)id * TM1 + t];
    mn = fminf(mn, e); mx = fmaxf(mx, e);
  }
  rmin2[id] = mn; rmax2[id] = mx;
}

__global__ void k_attn(const float* es, const float* rmin1, const float* rmax1,
                       const float* rmin2, const float* rmax2, float* attn) {
  const int b = threadIdx.x;
  if (b >= NB) return;
  float mx = -1e30f;
  for (int v = 0; v < NV; ++v)
    for (int t = 0; t < TM1; ++t) {
      const float e = es[((size_t)b * NV + v) * TM1 + t];
      const float eh = (e - rmin1[b * TM1 + t]) / (rmax1[b * TM1 + t] - rmin1[b * TM1 + t] + FEPS)
                     * (e - rmin2[b * NV + v]) / (rmax2[b * NV + v] - rmin2[b * NV + v] + FEPS)
                     * FSHARP;
      mx = fmaxf(mx, eh);
    }
  float sum = 0.f;
  for (int v = 0; v < NV; ++v)
    for (int t = 0; t < TM1; ++t) {
      const float e = es[((size_t)b * NV + v) * TM1 + t];
      const float eh = (e - rmin1[b * TM1 + t]) / (rmax1[b * TM1 + t] - rmin1[b * TM1 + t] + FEPS)
                     * (e - rmin2[b * NV + v]) / (rmax2[b * NV + v] - rmin2[b * NV + v] + FEPS)
                     * FSHARP;
      sum += expf(eh - mx);
    }
  for (int v = 0; v < NV; ++v)
    for (int t = 0; t < TM1; ++t) {
      const float e = es[((size_t)b * NV + v) * TM1 + t];
      const float eh = (e - rmin1[b * TM1 + t]) / (rmax1[b * TM1 + t] - rmin1[b * TM1 + t] + FEPS)
                     * (e - rmin2[b * NV + v]) / (rmax2[b * NV + v] - rmin2[b * NV + v] + FEPS)
                     * FSHARP;
      attn[((size_t)b * NV + v) * TM1 + t] = expf(eh - mx) / sum;
    }
}

__global__ void k_ctx(const float* hm, const float* attn, float* ctx) {
  const int id = blockIdx.x * blockDim.x + threadIdx.x;   // (b,v,d)
  if (id >= NB * NV * ND) return;
  const int b = id / (NV * ND);
  const int r = id % (NV * ND);
  const int v = r / ND, d = r % ND;
  float acc = 0.f;
  for (int t = 0; t < TM1; ++t)
    acc += attn[((size_t)b * NV + v) * TM1 + t] *
           hm[(((size_t)v * NB + b) * NT + t) * ND + d];
  ctx[id] = acc;
}

// conv f1: inputs cat[b,ci,v,:] where ci=0 -> last row of hm, ci=1 -> ctx
__global__ void k_conv1(const float* hm, const float* ctx,
                        const float* f1_w, const float* f1_b, float* y1) {
  const int id = blockIdx.x * blockDim.x + threadIdx.x;   // (b,o,v,p)
  if (id >= NB * 16 * NV * 480) return;
  const int b = id / (16 * NV * 480);
  int r = id % (16 * NV * 480);
  const int o = r / (NV * 480); r %= (NV * 480);
  const int vv = r / 480, p = r % 480;
  float acc = f1_b[o];
  const float* last = hm + (((size_t)vv * NB + b) * NT + (NT - 1)) * ND;
  const float* cx = ctx + ((size_t)b * NV + vv) * ND;
  for (int q = 0; q < 33; ++q) {
    acc += last[p + q] * f1_w[(o * 2 + 0) * 33 + q];
    acc += cx[p + q]   * f1_w[(o * 2 + 1) * 33 + q];
  }
  y1[id] = acc;
}

__global__ void k_bnstat1(const float* y1, float* m, float* v) {
  const int o = blockIdx.x, tid = threadIdx.x;
  float s = 0.f, ss = 0.f;
  for (int n = tid; n < NB * NV * 480; n += 256) {
    const int b = n / (NV * 480), r = n % (NV * 480);
    const float x = y1[((size_t)b * 16 + o) * (NV * 480) + r];
    s += x; ss += x * x;
  }
  __shared__ float S[256], SS[256];
  S[tid] = s; SS[tid] = ss; __syncthreads();
  for (int st = 128; st > 0; st >>= 1) {
    if (tid < st) { S[tid] += S[tid + st]; SS[tid] += SS[tid + st]; }
    __syncthreads();
  }
  if (tid == 0) {
    const float N = (float)(NB * NV * 480);
    const float mu = S[0] / N;
    m[o] = mu; v[o] = SS[0] / N - mu * mu;
  }
}

__global__ void k_bnapply1(float* y1, const float* m, const float* v,
                           const float* g, const float* bb) {
  const int id = blockIdx.x * blockDim.x + threadIdx.x;
  if (id >= NB * 16 * NV * 480) return;
  const int o = (id / (NV * 480)) % 16;
  const float x = (y1[id] - m[o]) * rsqrtf(v[o] + FEPS) * g[o] + bb[o];
  y1[id] = fmaxf(x, 0.f);
}

__global__ void k_conv2(const float* y1, const float* f2_w, const float* f2_b,
                        float* y2) {
  const int id = blockIdx.x * blockDim.x + threadIdx.x;   // (b,o2,p2)
  if (id >= NB * 32 * 478) return;
  const int b = id / (32 * 478);
  const int r = id % (32 * 478);
  const int o2 = r / 478, p2 = r % 478;
  float acc = f2_b[o2];
  for (int o = 0; o < 16; ++o)
    for (int vv = 0; vv < NV; ++vv) {
      const float* src = y1 + (((size_t)b * 16 + o) * NV + vv) * 480 + p2;
      const float* wq = f2_w + (((size_t)o2 * 16 + o) * NV + vv) * 3;
      acc += src[0] * wq[0] + src[1] * wq[1] + src[2] * wq[2];
    }
  y2[id] = acc;
}

__global__ void k_bnstat2(const float* y2, float* m, float* v) {
  const int o = blockIdx.x, tid = threadIdx.x;
  float s = 0.f, ss = 0.f;
  for (int n = tid; n < NB * 478; n += 256) {
    const int b = n / 478, p = n % 478;
    const float x = y2[((size_t)b * 32 + o) * 478 + p];
    s += x; ss += x * x;
  }
  __shared__ float S[256], SS[256];
  S[tid] = s; SS[tid] = ss; __syncthreads();
  for (int st = 128; st > 0; st >>= 1) {
    if (tid < st) { S[tid] += S[tid + st]; SS[tid] += SS[tid + st]; }
    __syncthreads();
  }
  if (tid == 0) {
    const float N = (float)(NB * 478);
    const float mu = S[0] / N;
    m[o] = mu; v[o] = SS[0] / N - mu * mu;
  }
}

__global__ void k_bnapply2(float* y2, const float* m, const float* v,
                           const float* g, const float* bb) {
  const int id = blockIdx.x * blockDim.x + threadIdx.x;
  if (id >= NB * 32 * 478) return;
  const int o = (id / 478) % 32;
  const float x = (y2[id] - m[o]) * rsqrtf(v[o] + FEPS) * g[o] + bb[o];
  y2[id] = fmaxf(x, 0.f);
}

__global__ void k_logit(const float* y2, const float* aff_w, const float* aff_b,
                        float* out) {
  const int id = threadIdx.x;
  if (id >= NB * 2) return;
  const int b = id >> 1, c = id & 1;
  float acc = aff_b[c];
  for (int o2 = 0; o2 < 32; ++o2)
    for (int p = 0; p < 478; ++p)
      acc += y2[((size_t)b * 32 + o2) * 478 + p] * aff_w[(size_t)c * 15296 + o2 * 478 + p];
  out[b * 2 + c] = acc;
}

// ================================================================
// Host launch
// ================================================================
extern "C" void kernel_launch(void* const* d_in, const int* in_sizes, int n_in,
                              void* d_out, int out_size, void* d_ws, size_t ws_size,
                              hipStream_t stream) {
  (void)in_sizes; (void)n_in; (void)out_size; (void)ws_size;
  const float* x       = (const float*)d_in[0];
  const float* l0f_wih = (const float*)d_in[1];
  const float* l0f_whh = (const float*)d_in[2];
  const float* l0f_bih = (const float*)d_in[3];
  const float* l0f_bhh = (const float*)d_in[4];
  const float* l0b_wih = (const float*)d_in[5];
  const float* l0b_whh = (const float*)d_in[6];
  const float* l0b_bih = (const float*)d_in[7];
  const float* l0b_bhh = (const float*)d_in[8];
  const float* l1f_wih = (const float*)d_in[9];
  const float* l1f_whh = (const float*)d_in[10];
  const float* l1f_bih = (const float*)d_in[11];
  const float* l1f_bhh = (const float*)d_in[12];
  const float* l1b_wih = (const float*)d_in[13];
  const float* l1b_whh = (const float*)d_in[14];
  const float* l1b_bih = (const float*)d_in[15];
  const float* l1b_bhh = (const float*)d_in[16];
  const float* sc_t_w = (const float*)d_in[17];
  const float* sc_t_b = (const float*)d_in[18];
  const float* sc_i_w = (const float*)d_in[19];
  const float* sc_i_b = (const float*)d_in[20];
  const float* cc1_w  = (const float*)d_in[21];
  const float* cc1_b  = (const float*)d_in[22];
  const float* cc2_w  = (const float*)d_in[23];
  const float* cc2_b  = (const float*)d_in[24];
  const float* wa_w   = (const float*)d_in[25];
  const float* wa_b   = (const float*)d_in[26];
  const float* wb_w   = (const float*)d_in[27];
  const float* wb_b   = (const float*)d_in[28];
  const float* wc_w   = (const float*)d_in[29];
  const float* wc_b   = (const float*)d_in[30];
  const float* wd_w   = (const float*)d_in[31];
  const float* wd_b   = (const float*)d_in[32];
  const float* f1_w   = (const float*)d_in[33];
  const float* f1_b   = (const float*)d_in[34];
  const float* bn1_g  = (const float*)d_in[35];
  const float* bn1_b  = (const float*)d_in[36];
  const float* f2_w   = (const float*)d_in[37];
  const float* f2_b   = (const float*)d_in[38];
  const float* bn2_g  = (const float*)d_in[39];
  const float* bn2_b  = (const float*)d_in[40];
  const float* aff_w  = (const float*)d_in[41];
  const float* aff_b  = (const float*)d_in[42];

  char* p = (char*)d_ws;
  auto alloc = [&](size_t bytes) -> char* {
    char* r = p;
    p += (bytes + 255) & ~(size_t)255;
    return r;
  };
  unsigned short* wcat0 = (unsigned short*)alloc((size_t)2 * 1024 * 256 * 2);
  unsigned short* wcat1 = (unsigned short*)alloc((size_t)2 * 1024 * 768 * 2);
  float*          bsum0 = (float*)alloc(2048 * 4);
  float*          bsum1 = (float*)alloc(2048 * 4);
  unsigned short* hbf0  = (unsigned short*)alloc((size_t)NV * NB * NT * ND * 2);
  unsigned short* hbf1  = (unsigned short*)alloc((size_t)NV * NB * NT * ND * 2);
  float*          hm32  = (float*)alloc((size_t)NV * NB * NT * ND * 4);
  float*          h_t   = (float*)alloc((size_t)NB * NK * 4);
  float*          h_tit = (float*)alloc((size_t)NB * NK * 4);
  float*          h_i   = (float*)alloc((size_t)NB * TM1 * NK * 4);
  float*          h_ti  = (float*)alloc((size_t)NB * TM1 * NK * 4);
  float*          soft  = (float*)alloc((size_t)NB * TM1 * NV * 4);
  float*          es    = (float*)alloc((size_t)NB * NV * TM1 * 4);
  float*          rmin1 = (float*)alloc((size_t)NB * TM1 * 4);
  float*          rmax1 = (float*)alloc((size_t)NB * TM1 * 4);
  float*          rmin2 = (float*)alloc((size_t)NB * NV * 4);
  float*          rmax2 = (float*)alloc((size_t)NB * NV * 4);
  float*          attn  = (float*)alloc((size_t)NB * NV * TM1 * 4);
  float*          ctx   = (float*)alloc((size_t)NB * NV * ND * 4);
  float*          y1    = (float*)alloc((size_t)NB * 16 * NV * 480 * 4);
  float*          y2    = (float*)alloc((size_t)NB * 32 * 478 * 4);
  float*          bn1m  = (float*)alloc(256);
  float*          bn1v  = (float*)alloc(256);
  float*          bn2m  = (float*)alloc(256);
  float*          bn2v  = (float*)alloc(256);

  // weight prep
  k_prep<<<6144, 256, 0, stream>>>(l0f_whh, l0b_whh, l1f_wih, l1f_whh, l1b_wih,
                                   l1b_whh, l0f_bih, l0f_bhh, l0b_bih, l0b_bhh,
                                   l1f_bih, l1f_bhh, l1b_bih, l1b_bhh,
                                   wcat0, wcat1, bsum0, bsum1);
  // LSTM layers (WMMA, fully fused, no global gate scratch)
  k_lstm<0><<<32, 512, 0, stream>>>(x, nullptr, wcat0, bsum0, l0f_wih, l0b_wih,
                                    hbf0, nullptr);
  k_lstm<512><<<32, 512, 0, stream>>>(nullptr, hbf0, wcat1, bsum1, nullptr, nullptr,
                                      hbf1, hm32);
  // tail
  k_ht<<<NB, 128, 0, stream>>>(hm32, sc_t_w, sc_t_b, cc1_w, cc1_b, h_t, h_tit);
  k_hi<<<NB * TM1, 128, 0, stream>>>(hm32, sc_i_w, sc_i_b, cc2_w, cc2_b, h_tit,
                                     h_i, h_ti);
  k_awsoft<<<(NB * TM1 + 3) / 4, 128, 0, stream>>>(hm32, soft);
  k_e<<<(NB * TM1 * NV) / 256, 256, 0, stream>>>(h_t, h_i, h_ti, soft, wa_w, wa_b,
                                                 wb_w, wb_b, wc_w, wc_b, wd_w, wd_b,
                                                 es);
  k_mm1<<<(NB * TM1 + 255) / 256, 256, 0, stream>>>(es, rmin1, rmax1);
  k_mm2<<<(NB * NV + 255) / 256, 256, 0, stream>>>(es, rmin2, rmax2);
  k_attn<<<1, 32, 0, stream>>>(es, rmin1, rmax1, rmin2, rmax2, attn);
  k_ctx<<<(NB * NV * ND) / 256, 256, 0, stream>>>(hm32, attn, ctx);
  k_conv1<<<(NB * 16 * NV * 480) / 256, 256, 0, stream>>>(hm32, ctx, f1_w, f1_b, y1);
  k_bnstat1<<<16, 256, 0, stream>>>(y1, bn1m, bn1v);
  k_bnapply1<<<(NB * 16 * NV * 480) / 256, 256, 0, stream>>>(y1, bn1m, bn1v, bn1_g, bn1_b);
  k_conv2<<<(NB * 32 * 478 + 255) / 256, 256, 0, stream>>>(y1, f2_w, f2_b, y2);
  k_bnstat2<<<32, 256, 0, stream>>>(y2, bn2m, bn2v);
  k_bnapply2<<<(NB * 32 * 478 + 255) / 256, 256, 0, stream>>>(y2, bn2m, bn2v, bn2_g, bn2_b);
  k_logit<<<1, 64, 0, stream>>>(y2, aff_w, aff_b, (float*)d_out);
}
